// se3_Encoder_48988396978552
// MI455X (gfx1250) — matrix operand order repaired and checked by reading, not compiled, not run on hardware
//
#include <hip/hip_runtime.h>
#include <math.h>

typedef __attribute__((ext_vector_type(16))) _Float16 v16h;
typedef __attribute__((ext_vector_type(8)))  float    v8f;
typedef __attribute__((ext_vector_type(4)))  unsigned v4u;
typedef __attribute__((ext_vector_type(8)))  int      v8i;
typedef __attribute__((ext_vector_type(4)))  int      v4i;
typedef _Float16 half_t;

#define NROWS   4096
#define IN_DIM  256
#define D_DIM   128
#define H_N     3
#define DH_DIM  128
#define L_DIM   512
#define B_DIM   8
#define RBF_N   16
#define ENC_N   100
#define OUT_N   32
#define CODES_N 40
#define AA_N    20
#define E_N     32768

#if defined(__has_builtin)
#  if __has_builtin(__builtin_amdgcn_tensor_load_to_lds) && __has_builtin(__builtin_amdgcn_s_wait_tensorcnt)
#    define HAVE_TDM 1
#  endif
#endif
#ifndef HAVE_TDM
#  define HAVE_TDM 0
#endif

static __device__ __forceinline__ float gelu_f(float x) {
  return 0.5f * x * (1.0f + erff(x * 0.70710678118654752f));
}

// ---------------------------------------------------------------------------
// Tensor Data Mover: DMA a 2D tile (rows x row_dwords DWORDs, row stride
// stride_dwords DWORDs) from global memory into LDS at lds_off.
// LDS padding: 2 DWORDs inserted after every 16 DWORDs stored (matches the
// [64][36] half_t LDS tiles below).  D# layout per CDNA5 ISA section 8.
// ---------------------------------------------------------------------------
static __device__ __forceinline__ void tdm_load_tile(unsigned lds_off, const void* gaddr,
                                                     int rows, int row_dwords,
                                                     long long stride_dwords)
{
#if HAVE_TDM
  unsigned long long ga = (unsigned long long)gaddr;
  v4u g0;
  g0[0] = 1u;                                             // count=1
  g0[1] = lds_off;                                        // lds_addr (bytes)
  g0[2] = (unsigned)(ga & 0xFFFFFFFFu);                   // global_addr[31:0]
  g0[3] = (unsigned)((ga >> 32) & 0x01FFFFFFu) | (2u << 30); // [56:32] | type=2
  unsigned td0 = 0x7FFFFFFFu, td1 = 0x7FFFFFFFu;          // huge tensor dims (no OOB)
  unsigned long long s0 = (unsigned long long)stride_dwords;
  v8i g1;
  g1[0] = (int)((2u << 16) | (1u << 20) | (3u << 22) | (1u << 25)); // 4B elems, pad 2dw/16dw
  g1[1] = (int)((td0 & 0xFFFFu) << 16);                   // tensor_dim0 lo
  g1[2] = (int)(((td0 >> 16) & 0xFFFFu) | ((td1 & 0xFFFFu) << 16));
  g1[3] = (int)(((td1 >> 16) & 0xFFFFu) | ((unsigned)row_dwords << 16)); // tile_dim0
  g1[4] = rows;                                           // tile_dim1 (tile_dim2=0)
  g1[5] = (int)(s0 & 0xFFFFFFFFu);                        // dim0_stride lo
  g1[6] = (int)((s0 >> 32) & 0xFFFFu);                    // dim0_stride hi (dim1_stride=0)
  g1[7] = 0;
  v4i z4 = {0, 0, 0, 0};
#if __clang_major__ >= 23
  v8i z8 = {0, 0, 0, 0, 0, 0, 0, 0};
  __builtin_amdgcn_tensor_load_to_lds(g0, g1, z4, z4, z8, 0);
#else
  __builtin_amdgcn_tensor_load_to_lds(g0, g1, z4, z4, 0);
#endif
#else
  (void)lds_off; (void)gaddr; (void)rows; (void)row_dwords; (void)stride_dwords;
#endif
}

static __device__ __forceinline__ void tdm_wait()
{
#if HAVE_TDM
  __builtin_amdgcn_s_wait_tensorcnt(0);
#endif
}

// ---------------------------------------------------------------------------
// General WMMA GEMM: D = act(alpha * A@B + bias)
// A: MxK f16 row-major (lda). B: KxN f16 row-major (ldb) or, if transB, NxK.
// Batched via blockIdx.z with element strides sA/sB/sD.
// Block 256 threads = 8 waves; block tile 64(M) x 64(N), BK=32.
// Each wave computes a 32x16 tile: 2 x v_wmma_f32_16x16x32_f16 per k-step.
// Interior tiles are staged into LDS by the Tensor Data Mover (wave 0 issues
// tensor_load_to_lds + s_wait_tensorcnt); ragged tiles use branchless clamped
// loads (no exec-mask divergence).
// ---------------------------------------------------------------------------
__global__ void __launch_bounds__(256)
gemm_wmma(const half_t* __restrict__ A, const half_t* __restrict__ Bm,
          const float* __restrict__ bias, float* __restrict__ Dm,
          int M, int N, int K, int lda, int ldb, int ldd,
          long long sA, long long sB, long long sD,
          float alpha, int transB, int act)
{
  const half_t* Ab = A  + (long long)blockIdx.z * sA;
  const half_t* Bb = Bm + (long long)blockIdx.z * sB;
  float*        Db = Dm + (long long)blockIdx.z * sD;

  __shared__ half_t As[64][36];   // [m][k], 8B pad per row
  __shared__ half_t Bs[64][36];   // [n][k], staged transposed

  const int tid  = threadIdx.x;
  const int wave = tid >> 5;
  const int lane = tid & 31;
  const int wr   = wave >> 2;       // 0..1 -> 32-row half
  const int wc   = wave & 3;        // 0..3 -> 16-col strip
  const int row0 = blockIdx.y * 64;
  const int col0 = blockIdx.x * 64;
  const int r    = lane & 15;
  const int hi   = lane >> 4;

  v8f acc0 = {};
  v8f acc1 = {};

  const unsigned ldsA = (unsigned)(size_t)&As[0][0];
  const unsigned ldsB = (unsigned)(size_t)&Bs[0][0];

  for (int k0 = 0; k0 < K; k0 += 32) {
    const bool fullK = (k0 + 32) <= K;
    const bool fastA = HAVE_TDM && fullK && (row0 + 64) <= M;
    const bool fastB = HAVE_TDM && fullK && transB && (col0 + 64) <= N;

    if (fastA || fastB) {
      if (wave == 0) {
        if (fastA) tdm_load_tile(ldsA, Ab + (long long)row0 * lda + k0, 64, 16, (long long)(lda >> 1));
        if (fastB) tdm_load_tile(ldsB, Bb + (long long)col0 * ldb + k0, 64, 16, (long long)(ldb >> 1));
        tdm_wait();
      }
    }
    if (!fastA) {
      // branchless clamped staging of A tile: 64x32, 8 halves / thread
      int ar   = tid >> 2;
      int ak   = (tid & 3) << 3;
      int grow = row0 + ar;
      int growc = grow < M ? grow : (M - 1);
      const half_t* src = Ab + (long long)growc * lda + k0;
      int kmax = K - k0;            // valid kk in [0,kmax)
      bool rok = grow < M;
#pragma unroll
      for (int j = 0; j < 8; ++j) {
        int kk  = ak + j;
        int kkc = kk < kmax ? kk : (kmax - 1);
        half_t v = src[kkc];
        As[ar][kk] = (rok && kk < kmax) ? v : (half_t)0.f;
      }
    }
    if (!fastB) {
      if (transB) {
        int n    = tid >> 2;
        int kk0  = (tid & 3) << 3;
        int gcol = col0 + n;
        int gc   = gcol < N ? gcol : (N - 1);
        const half_t* src = Bb + (long long)gc * ldb + k0;
        int kmax = K - k0;
        bool cok = gcol < N;
#pragma unroll
        for (int j = 0; j < 8; ++j) {
          int kk  = kk0 + j;
          int kkc = kk < kmax ? kk : (kmax - 1);
          half_t v = src[kkc];
          Bs[n][kk] = (cok && kk < kmax) ? v : (half_t)0.f;
        }
      } else {
        int kk = tid >> 3;            // 0..31
        int n0 = (tid & 7) << 3;
        int gk = k0 + kk;
        int kc = gk < K ? kk : (K - 1 - k0);
        const half_t* src = Bb + (long long)(k0 + kc) * ldb + col0;
        bool kok = gk < K;
#pragma unroll
        for (int j = 0; j < 8; ++j) {
          int n  = n0 + j;
          int nc = (col0 + n) < N ? n : (N - 1 - col0);
          half_t v = src[nc];
          Bs[n][kk] = (kok && (col0 + n) < N) ? v : (half_t)0.f;
        }
      }
    }
    __syncthreads();

    // fragments per CDNA5 ISA 7.12.2 layouts
    v16h a0, a1, bf;
#pragma unroll
    for (int j = 0; j < 8; ++j) {
      a0[j]     = As[wr * 32 + r][hi * 8 + j];
      a0[8 + j] = As[wr * 32 + r][16 + hi * 8 + j];
      a1[j]     = As[wr * 32 + 16 + r][hi * 8 + j];
      a1[8 + j] = As[wr * 32 + 16 + r][16 + hi * 8 + j];
    }
#pragma unroll
    for (int j = 0; j < 8; ++j) { bf[j] = Bs[wc * 16 + r][hi * 8 + j]; bf[8 + j] = Bs[wc * 16 + r][16 + hi * 8 + j]; }

    acc0 = __builtin_amdgcn_wmma_f32_16x16x32_f16(false, a0, false, bf,
                                                  (short)0, acc0, false, false);
    acc1 = __builtin_amdgcn_wmma_f32_16x16x32_f16(false, a1, false, bf,
                                                  (short)0, acc1, false, false);
    __syncthreads();
  }

  // store: C/D layout: VGPR rr -> M = hi*8+rr, N = lane&15
  int col = col0 + wc * 16 + r;
  if (col < N) {
    float bv = bias ? bias[col] : 0.f;
#pragma unroll
    for (int rr = 0; rr < 8; ++rr) {
      int row = row0 + wr * 32 + hi * 8 + rr;
      if (row < M) {
        float v = acc0[rr] * alpha + bv;
        if (act == 1) v = gelu_f(v);
        Db[(long long)row * ldd + col] = v;
      }
      int row2 = row + 16;
      if (row2 < M) {
        float v = acc1[rr] * alpha + bv;
        if (act == 1) v = gelu_f(v);
        Db[(long long)row2 * ldd + col] = v;
      }
    }
  }
}

// ---------------------------------------------------------------------------
// Elementwise / reduction kernels
// ---------------------------------------------------------------------------
__global__ void bn_partial_kernel(const float* __restrict__ x,
                                  float* __restrict__ gsum, float* __restrict__ gsq)
{
  int f  = threadIdx.x;            // 256 features
  int r0 = blockIdx.x * 64;        // 64 rows per block, 64 blocks
  float s = 0.f, q = 0.f;
  for (int r = 0; r < 64; ++r) {
    float v = x[(long long)(r0 + r) * IN_DIM + f];
    s += v; q += v * v;
  }
  atomicAdd(&gsum[f], s);
  atomicAdd(&gsq[f], q);
}

__global__ void bn_apply_kernel(const float* __restrict__ x,
                                const float* __restrict__ gsum, const float* __restrict__ gsq,
                                const float* __restrict__ g, const float* __restrict__ b,
                                half_t* __restrict__ out)
{
  long long idx = (long long)blockIdx.x * 256 + threadIdx.x;
  if (idx >= (long long)NROWS * IN_DIM) return;
  int f = (int)(idx & (IN_DIM - 1));
  float m = gsum[f] * (1.f / NROWS);
  float v = gsq[f] * (1.f / NROWS) - m * m;
  float y = (x[idx] - m) * rsqrtf(v + 1e-5f) * g[f] + b[f];
  out[idx] = (half_t)y;
}

__global__ void f2h_kernel(const float* __restrict__ in, half_t* __restrict__ out, long long n)
{
  long long idx = (long long)blockIdx.x * 256 + threadIdx.x;
  if (idx < n) out[idx] = (half_t)in[idx];
}

__global__ void add_kernel(float* __restrict__ x, const float* __restrict__ y, long long n)
{
  long long idx = (long long)blockIdx.x * 256 + threadIdx.x;
  if (idx < n) x[idx] += y[idx];
}

__global__ void dyt_kernel(const float* __restrict__ in, float* __restrict__ out,
                           const float* __restrict__ a, const float* __restrict__ w,
                           const float* __restrict__ b, int F, long long n)
{
  long long idx = (long long)blockIdx.x * 256 + threadIdx.x;
  if (idx >= n) return;
  int f = (int)(idx % F);
  out[idx] = tanhf(a[0] * in[idx]) * w[f] + b[f];
}

__global__ void ln_kernel(const float* __restrict__ x, const float* __restrict__ g,
                          const float* __restrict__ b, half_t* __restrict__ out)
{
  int row = blockIdx.x;        // one of 4096 rows, D=128
  int t   = threadIdx.x;       // 128 threads
  __shared__ float red[128];
  float v = x[(long long)row * D_DIM + t];
  red[t] = v; __syncthreads();
  for (int s = 64; s > 0; s >>= 1) { if (t < s) red[t] += red[t + s]; __syncthreads(); }
  float mean = red[0] * (1.f / D_DIM);
  __syncthreads();
  float d = v - mean;
  red[t] = d * d; __syncthreads();
  for (int s = 64; s > 0; s >>= 1) { if (t < s) red[t] += red[t + s]; __syncthreads(); }
  float var = red[0] * (1.f / D_DIM);
  float y = d * rsqrtf(var + 1e-5f) * g[t] + b[t];
  out[(long long)row * D_DIM + t] = (half_t)y;
}

// (N,H,DH) f32  ->  (B,H,L,DH) f16   (for batched attention GEMMs)
__global__ void qkvperm_kernel(const float* __restrict__ in, half_t* __restrict__ out, long long n)
{
  long long idx = (long long)blockIdx.x * 256 + threadIdx.x;
  if (idx >= n) return;
  long long nn = idx / (H_N * DH_DIM);
  int hf = (int)(idx - nn * (H_N * DH_DIM));
  int h = hf >> 7, f = hf & 127;
  long long b = nn >> 9, i = nn & 511;
  out[(((b * H_N + h) * L_DIM + i) * DH_DIM) + f] = (half_t)in[idx];
}

// (B,H,L,DH) f32 -> (N, H*DH) f16   (attention output for the Wo GEMM)
__global__ void ohperm_kernel(const float* __restrict__ in, half_t* __restrict__ out, long long n)
{
  long long idx = (long long)blockIdx.x * 256 + threadIdx.x;
  if (idx >= n) return;
  long long nn = idx / (H_N * DH_DIM);
  int hf = (int)(idx - nn * (H_N * DH_DIM));
  int h = hf >> 7, f = hf & 127;
  long long b = nn >> 9, i = nn & 511;
  out[idx] = (half_t)in[(((b * H_N + h) * L_DIM + i) * DH_DIM) + f];
}

__global__ void edge_scatter_kernel(const int* __restrict__ ei, unsigned* __restrict__ adjw)
{
  int e = blockIdx.x * 256 + threadIdx.x;
  if (e >= E_N) return;
  int r = ei[e];
  int c = ei[E_N + e];
  if ((r >> 9) == (c >> 9)) {
    unsigned b = (unsigned)(r >> 9), i = (unsigned)(r & 511), j = (unsigned)(c & 511);
    unsigned idx = (b << 18) | (i << 9) | j;
    atomicOr(&adjw[idx >> 5], 1u << (idx & 31));
  }
}

// Fused RBF distance bias + adjacency mask over logits S laid out (B*H, L, L)
__global__ void rbf_mask_kernel(float* __restrict__ S, const float* __restrict__ coords,
                                const float* __restrict__ wr, const unsigned* __restrict__ adjw)
{
  long long idx = (long long)blockIdx.x * 256 + threadIdx.x;
  if (idx >= (long long)B_DIM * L_DIM * L_DIM) return;
  int b = (int)(idx >> 18);
  int rem = (int)(idx & 262143);
  int i = rem >> 9, j = rem & 511;
  long long base = ((long long)(b * H_N) * L_DIM + i) * L_DIM + j;
  int bit = (adjw[idx >> 5] >> (idx & 31)) & 1;
  if (!bit) {
    S[base] = -1e9f; S[base + 262144] = -1e9f; S[base + 2 * 262144] = -1e9f;
    return;
  }
  const float* pi = coords + (long long)(b * L_DIM + i) * 3;
  const float* pj = coords + (long long)(b * L_DIM + j) * 3;
  float dx = pi[0] - pj[0], dy = pi[1] - pj[1], dz = pi[2] - pj[2];
  float dist = sqrtf(dx * dx + dy * dy + dz * dz + 1e-8f);
  float rb[RBF_N];
  const float step = 10.f / (RBF_N - 1);
  const float inv2s2 = 1.f / (2.f * (10.f / RBF_N) * (10.f / RBF_N));
#pragma unroll
  for (int rr = 0; rr < RBF_N; ++rr) {
    float d = dist - step * rr;
    rb[rr] = __expf(-d * d * inv2s2);
  }
#pragma unroll
  for (int h = 0; h < H_N; ++h) {
    float a = 0.f;
#pragma unroll
    for (int rr = 0; rr < RBF_N; ++rr) a += rb[rr] * wr[rr * H_N + h];
    S[base + (long long)h * 262144] += a;
  }
}

// Row softmax over L=512 with hard-mask zeroing; writes f16 probs
__global__ void softmax_kernel(const float* __restrict__ S, half_t* __restrict__ P)
{
  int row = blockIdx.x;          // (b*H+h)*L + i
  int t = threadIdx.x;           // 128 threads, 4 elems each
  const float* s = S + (long long)row * L_DIM;
  half_t* p = P + (long long)row * L_DIM;
  __shared__ float red[128];
  float vals[4], mx = -3.4e38f;
#pragma unroll
  for (int j = 0; j < 4; ++j) { vals[j] = s[t + j * 128]; mx = fmaxf(mx, vals[j]); }
  red[t] = mx; __syncthreads();
  for (int st = 64; st > 0; st >>= 1) { if (t < st) red[t] = fmaxf(red[t], red[t + st]); __syncthreads(); }
  mx = red[0]; __syncthreads();
  float e[4], sum = 0.f;
#pragma unroll
  for (int j = 0; j < 4; ++j) {
    e[j] = (vals[j] < -1e8f) ? 0.f : __expf(vals[j] - mx);
    sum += e[j];
  }
  red[t] = sum; __syncthreads();
  for (int st = 64; st > 0; st >>= 1) { if (t < st) red[t] += red[t + st]; __syncthreads(); }
  sum = red[0];
  float inv = sum > 0.f ? 1.f / sum : 0.f;
#pragma unroll
  for (int j = 0; j < 4; ++j) p[t + j * 128] = (half_t)(e[j] * inv);
}

__global__ void concat_kernel(const float* __restrict__ u2, const float* __restrict__ aa,
                              half_t* __restrict__ cat, long long n)
{
  long long idx = (long long)blockIdx.x * 256 + threadIdx.x;
  if (idx >= n) return;
  long long r = idx / (ENC_N + AA_N);
  int c = (int)(idx - r * (ENC_N + AA_N));
  float v = (c < ENC_N) ? u2[r * ENC_N + c] : aa[r * AA_N + (c - ENC_N)];
  cat[idx] = (half_t)v;
}

__global__ void tailact_kernel(const float* __restrict__ z0, float* __restrict__ z,
                               const float* __restrict__ a, const float* __restrict__ w,
                               const float* __restrict__ b, long long n)
{
  long long idx = (long long)blockIdx.x * 256 + threadIdx.x;
  if (idx >= n) return;
  int f = (int)(idx & (OUT_N - 1));
  float x = gelu_f(z0[idx]);
  x = tanhf(a[0] * x) * w[f] + b[f];
  z[idx] = tanhf(x);
}

__global__ void vq_kernel(const float* __restrict__ z, const float* __restrict__ codes,
                          float* __restrict__ out, float* __restrict__ acc)
{
  int i = blockIdx.x * 256 + threadIdx.x;
  if (i >= NROWS) return;
  float zr[OUT_N];
#pragma unroll
  for (int j = 0; j < OUT_N; ++j) zr[j] = z[(long long)i * OUT_N + j];
  float best = 3.4e38f; int bi = 0;
  for (int c = 0; c < CODES_N; ++c) {
    float d = 0.f;
#pragma unroll
    for (int j = 0; j < OUT_N; ++j) { float t = zr[j] - codes[c * OUT_N + j]; d += t * t; }
    if (d < best) { best = d; bi = c; }
  }
  float s = 0.f;
#pragma unroll
  for (int j = 0; j < OUT_N; ++j) {
    float e = codes[bi * OUT_N + j];
    out[(long long)i * OUT_N + j] = e;      // zq = z + sg(e - z) == e
    float df = e - zr[j];
    s += df * df;
  }
  atomicAdd(acc, s);
}

__global__ void loss_kernel(float* __restrict__ out_loss, const float* __restrict__ acc)
{
  if (threadIdx.x == 0)
    out_loss[0] = 1.25f * acc[0] / (float)(NROWS * OUT_N);  // q_loss + 0.25*e_loss
}

// ---------------------------------------------------------------------------
extern "C" void kernel_launch(void* const* d_in, const int* in_sizes, int n_in,
                              void* d_out, int out_size, void* d_ws, size_t ws_size,
                              hipStream_t stream)
{
  (void)in_sizes; (void)out_size;
  if (n_in < 54) return;

  const float* x_res  = (const float*)d_in[0];
  const float* coords = (const float*)d_in[1];
  const float* aa     = (const float*)d_in[2];
  const int*   eidx   = (const int*)  d_in[3];
  const float* bn_g   = (const float*)d_in[4];
  const float* bn_b   = (const float*)d_in[5];
  const float* ffin_w1 = (const float*)d_in[6];
  const float* ffin_b1 = (const float*)d_in[7];
  const float* ffin_w2 = (const float*)d_in[8];
  const float* ffin_b2 = (const float*)d_in[9];
  const float* ffin_dyt_a = (const float*)d_in[10];
  const float* ffin_dyt_w = (const float*)d_in[11];
  const float* ffin_dyt_b = (const float*)d_in[12];
  struct Lp { const float *ln1_g,*ln1_b,*wq,*wk,*wv,*wr,*wo,*bo,*ln2_g,*ln2_b,*wf1,*bf1,*wf2,*bf2; } lp[2];
  for (int l = 0; l < 2; ++l) {
    int o = 13 + l * 14;
    lp[l].ln1_g = (const float*)d_in[o+0];  lp[l].ln1_b = (const float*)d_in[o+1];
    lp[l].wq    = (const float*)d_in[o+2];  lp[l].wk    = (const float*)d_in[o+3];
    lp[l].wv    = (const float*)d_in[o+4];  lp[l].wr    = (const float*)d_in[o+5];
    lp[l].wo    = (const float*)d_in[o+6];  lp[l].bo    = (const float*)d_in[o+7];
    lp[l].ln2_g = (const float*)d_in[o+8];  lp[l].ln2_b = (const float*)d_in[o+9];
    lp[l].wf1   = (const float*)d_in[o+10]; lp[l].bf1   = (const float*)d_in[o+11];
    lp[l].wf2   = (const float*)d_in[o+12]; lp[l].bf2   = (const float*)d_in[o+13];
  }
  const float* lin_dyt_a = (const float*)d_in[41];
  const float* lin_dyt_w = (const float*)d_in[42];
  const float* lin_dyt_b = (const float*)d_in[43];
  const float* lin_w1 = (const float*)d_in[44];
  const float* lin_b1 = (const float*)d_in[45];
  const float* lin_w2 = (const float*)d_in[46];
  const float* lin_b2 = (const float*)d_in[47];
  const float* out_w  = (const float*)d_in[48];
  const float* out_b  = (const float*)d_in[49];
  const float* out_dyt_a = (const float*)d_in[50];
  const float* out_dyt_w = (const float*)d_in[51];
  const float* out_dyt_b = (const float*)d_in[52];
  const float* codebook  = (const float*)d_in[53];

  // ---- workspace bump allocator --------------------------------------------
  char* wp = (char*)d_ws;
  auto alloc = [&](size_t bytes) -> void* {
    void* p = (void*)wp;
    wp += (bytes + 255) & ~(size_t)255;
    return p;
  };
  float*    gsum  = (float*)alloc(IN_DIM * 4);
  float*    gsq   = (float*)alloc(IN_DIM * 4);
  float*    vqacc = (float*)alloc(256);
  unsigned* adjw  = (unsigned*)alloc((size_t)B_DIM * L_DIM * L_DIM / 8);
  size_t zero_bytes = (size_t)(wp - (char*)d_ws);

  half_t* x256h = (half_t*)alloc((size_t)NROWS * IN_DIM * 2);
  float*  t0f   = (float*) alloc((size_t)NROWS * IN_DIM * 4);
  half_t* t0h   = (half_t*)alloc((size_t)NROWS * IN_DIM * 2);
  float*  xf    = (float*) alloc((size_t)NROWS * D_DIM * 4);
  half_t* hh    = (half_t*)alloc((size_t)NROWS * D_DIM * 2);
  float*  qf    = (float*) alloc((size_t)NROWS * H_N * DH_DIM * 4);
  half_t* q_h   = (half_t*)alloc((size_t)NROWS * H_N * DH_DIM * 2);
  half_t* k_h   = (half_t*)alloc((size_t)NROWS * H_N * DH_DIM * 2);
  half_t* v_h   = (half_t*)alloc((size_t)NROWS * H_N * DH_DIM * 2);
  float*  Sbuf  = (float*) alloc((size_t)B_DIM * H_N * L_DIM * L_DIM * 4);
  half_t* p_h   = (half_t*)alloc((size_t)B_DIM * H_N * L_DIM * L_DIM * 2);
  float*  of    = (float*) alloc((size_t)NROWS * H_N * DH_DIM * 4);
  half_t* o_h   = (half_t*)alloc((size_t)NROWS * H_N * DH_DIM * 2);
  float*  projf = (float*) alloc((size_t)NROWS * D_DIM * 4);
  float*  ff512 = (float*) alloc((size_t)NROWS * 4 * D_DIM * 4);
  half_t* f_h   = (half_t*)alloc((size_t)NROWS * 4 * D_DIM * 2);
  float*  tf    = (float*) alloc((size_t)NROWS * D_DIM * 4);
  half_t* t_h   = (half_t*)alloc((size_t)NROWS * D_DIM * 2);
  float*  uf    = (float*) alloc((size_t)NROWS * ENC_N * 4);
  half_t* u_h   = (half_t*)alloc((size_t)NROWS * ENC_N * 2);
  float*  u2f   = (float*) alloc((size_t)NROWS * ENC_N * 4);
  half_t* cat_h = (half_t*)alloc((size_t)NROWS * (ENC_N + AA_N) * 2);
  float*  z0f   = (float*) alloc((size_t)NROWS * OUT_N * 4);
  float*  zf    = (float*) alloc((size_t)NROWS * OUT_N * 4);
  half_t* ffin_w1h = (half_t*)alloc((size_t)IN_DIM * 2 * D_DIM * 2);
  half_t* ffin_w2h = (half_t*)alloc((size_t)2 * D_DIM * D_DIM * 2);
  half_t *wq_h[2], *wk_h[2], *wv_h[2], *wo_h[2], *wf1_h[2], *wf2_h[2];
  for (int l = 0; l < 2; ++l) {
    wq_h[l]  = (half_t*)alloc((size_t)D_DIM * H_N * DH_DIM * 2);
    wk_h[l]  = (half_t*)alloc((size_t)D_DIM * H_N * DH_DIM * 2);
    wv_h[l]  = (half_t*)alloc((size_t)D_DIM * H_N * DH_DIM * 2);
    wo_h[l]  = (half_t*)alloc((size_t)H_N * DH_DIM * D_DIM * 2);
    wf1_h[l] = (half_t*)alloc((size_t)D_DIM * 4 * D_DIM * 2);
    wf2_h[l] = (half_t*)alloc((size_t)4 * D_DIM * D_DIM * 2);
  }
  half_t* lin_w1h = (half_t*)alloc((size_t)D_DIM * ENC_N * 2);
  half_t* lin_w2h = (half_t*)alloc((size_t)ENC_N * ENC_N * 2);
  half_t* out_wh  = (half_t*)alloc((size_t)(ENC_N + AA_N) * OUT_N * 2);
  if ((size_t)(wp - (char*)d_ws) > ws_size) return;   // not enough scratch

  // ---- helpers -------------------------------------------------------------
  auto gemm = [&](const half_t* A, const half_t* Bm, const float* bias, float* Dm,
                  int M, int N, int K, int lda, int ldb, int ldd,
                  long long sA, long long sB, long long sD, int nb,
                  float alpha, int transB, int act) {
    dim3 g((N + 63) / 64, (M + 63) / 64, nb);
    gemm_wmma<<<g, 256, 0, stream>>>(A, Bm, bias, Dm, M, N, K, lda, ldb, ldd,
                                     sA, sB, sD, alpha, transB, act);
  };
  auto f2h = [&](const float* in, half_t* out, long long n) {
    f2h_kernel<<<(int)((n + 255) / 256), 256, 0, stream>>>(in, out, n);
  };

  // ---- zero reduction buffers + adjacency bitmask --------------------------
  hipMemsetAsync(d_ws, 0, zero_bytes, stream);

  // ---- weights to f16 ------------------------------------------------------
  f2h(ffin_w1, ffin_w1h, (long long)IN_DIM * 2 * D_DIM);
  f2h(ffin_w2, ffin_w2h, (long long)2 * D_DIM * D_DIM);
  for (int l = 0; l < 2; ++l) {
    f2h(lp[l].wq,  wq_h[l],  (long long)D_DIM * H_N * DH_DIM);
    f2h(lp[l].wk,  wk_h[l],  (long long)D_DIM * H_N * DH_DIM);
    f2h(lp[l].wv,  wv_h[l],  (long long)D_DIM * H_N * DH_DIM);
    f2h(lp[l].wo,  wo_h[l],  (long long)H_N * DH_DIM * D_DIM);
    f2h(lp[l].wf1, wf1_h[l], (long long)D_DIM * 4 * D_DIM);
    f2h(lp[l].wf2, wf2_h[l], (long long)4 * D_DIM * D_DIM);
  }
  f2h(lin_w1, lin_w1h, (long long)D_DIM * ENC_N);
  f2h(lin_w2, lin_w2h, (long long)ENC_N * ENC_N);
  f2h(out_w,  out_wh,  (long long)(ENC_N + AA_N) * OUT_N);

  // ---- batchnorm + input MLP ----------------------------------------------
  bn_partial_kernel<<<64, 256, 0, stream>>>(x_res, gsum, gsq);
  bn_apply_kernel<<<(NROWS * IN_DIM + 255) / 256, 256, 0, stream>>>(x_res, gsum, gsq, bn_g, bn_b, x256h);
  gemm(x256h, ffin_w1h, ffin_b1, t0f, NROWS, 2 * D_DIM, IN_DIM, IN_DIM, 2 * D_DIM, 2 * D_DIM,
       0, 0, 0, 1, 1.f, 0, 1);
  f2h(t0f, t0h, (long long)NROWS * 2 * D_DIM);
  gemm(t0h, ffin_w2h, ffin_b2, xf, NROWS, D_DIM, 2 * D_DIM, 2 * D_DIM, D_DIM, D_DIM,
       0, 0, 0, 1, 1.f, 0, 1);
  dyt_kernel<<<(NROWS * D_DIM + 255) / 256, 256, 0, stream>>>(
      xf, xf, ffin_dyt_a, ffin_dyt_w, ffin_dyt_b, D_DIM, (long long)NROWS * D_DIM);

  // ---- adjacency mask ------------------------------------------------------
  edge_scatter_kernel<<<(E_N + 255) / 256, 256, 0, stream>>>(eidx, adjw);

  // ---- transformer layers --------------------------------------------------
  const float scale = 0.08838834764831845f;  // 1/sqrt(128)
  const long long NHD = (long long)NROWS * H_N * DH_DIM;
  for (int l = 0; l < 2; ++l) {
    ln_kernel<<<NROWS, 128, 0, stream>>>(xf, lp[l].ln1_g, lp[l].ln1_b, hh);
    // Q, K, V projections (D -> H*DH), then permute to (B,H,L,DH) f16
    gemm(hh, wq_h[l], nullptr, qf, NROWS, H_N * DH_DIM, D_DIM, D_DIM, H_N * DH_DIM, H_N * DH_DIM,
         0, 0, 0, 1, 1.f, 0, 0);
    qkvperm_kernel<<<(int)((NHD + 255) / 256), 256, 0, stream>>>(qf, q_h, NHD);
    gemm(hh, wk_h[l], nullptr, qf, NROWS, H_N * DH_DIM, D_DIM, D_DIM, H_N * DH_DIM, H_N * DH_DIM,
         0, 0, 0, 1, 1.f, 0, 0);
    qkvperm_kernel<<<(int)((NHD + 255) / 256), 256, 0, stream>>>(qf, k_h, NHD);
    gemm(hh, wv_h[l], nullptr, qf, NROWS, H_N * DH_DIM, D_DIM, D_DIM, H_N * DH_DIM, H_N * DH_DIM,
         0, 0, 0, 1, 1.f, 0, 0);
    qkvperm_kernel<<<(int)((NHD + 255) / 256), 256, 0, stream>>>(qf, v_h, NHD);
    // logits S = scale * Q K^T, batched over B*H=24 (transB path -> TDM-staged)
    gemm(q_h, k_h, nullptr, Sbuf, L_DIM, L_DIM, DH_DIM, DH_DIM, DH_DIM, L_DIM,
         (long long)L_DIM * DH_DIM, (long long)L_DIM * DH_DIM, (long long)L_DIM * L_DIM,
         B_DIM * H_N, scale, 1, 0);
    // + RBF distance bias, adjacency mask
    rbf_mask_kernel<<<(int)(((long long)B_DIM * L_DIM * L_DIM + 255) / 256), 256, 0, stream>>>(
        Sbuf, coords, lp[l].wr, adjw);
    softmax_kernel<<<B_DIM * H_N * L_DIM, 128, 0, stream>>>(Sbuf, p_h);
    // O = P V, batched over 24
    gemm(p_h, v_h, nullptr, of, L_DIM, DH_DIM, L_DIM, L_DIM, DH_DIM, DH_DIM,
         (long long)L_DIM * L_DIM, (long long)L_DIM * DH_DIM, (long long)L_DIM * DH_DIM,
         B_DIM * H_N, 1.f, 0, 0);
    ohperm_kernel<<<(int)((NHD + 255) / 256), 256, 0, stream>>>(of, o_h, NHD);
    // output projection + residual
    gemm(o_h, wo_h[l], lp[l].bo, projf, NROWS, D_DIM, H_N * DH_DIM, H_N * DH_DIM, D_DIM, D_DIM,
         0, 0, 0, 1, 1.f, 0, 0);
    add_kernel<<<(NROWS * D_DIM + 255) / 256, 256, 0, stream>>>(xf, projf, (long long)NROWS * D_DIM);
    // FFN
    ln_kernel<<<NROWS, 128, 0, stream>>>(xf, lp[l].ln2_g, lp[l].ln2_b, hh);
    gemm(hh, wf1_h[l], lp[l].bf1, ff512, NROWS, 4 * D_DIM, D_DIM, D_DIM, 4 * D_DIM, 4 * D_DIM,
         0, 0, 0, 1, 1.f, 0, 1);
    f2h(ff512, f_h, (long long)NROWS * 4 * D_DIM);
    gemm(f_h, wf2_h[l], lp[l].bf2, projf, NROWS, D_DIM, 4 * D_DIM, 4 * D_DIM, D_DIM, D_DIM,
         0, 0, 0, 1, 1.f, 0, 0);
    add_kernel<<<(NROWS * D_DIM + 255) / 256, 256, 0, stream>>>(xf, projf, (long long)NROWS * D_DIM);
  }

  // ---- tail MLP + VQ -------------------------------------------------------
  dyt_kernel<<<(NROWS * D_DIM + 255) / 256, 256, 0, stream>>>(
      xf, tf, lin_dyt_a, lin_dyt_w, lin_dyt_b, D_DIM, (long long)NROWS * D_DIM);
  f2h(tf, t_h, (long long)NROWS * D_DIM);
  gemm(t_h, lin_w1h, lin_b1, uf, NROWS, ENC_N, D_DIM, D_DIM, ENC_N, ENC_N,
       0, 0, 0, 1, 1.f, 0, 1);
  f2h(uf, u_h, (long long)NROWS * ENC_N);
  gemm(u_h, lin_w2h, lin_b2, u2f, NROWS, ENC_N, ENC_N, ENC_N, ENC_N, ENC_N,
       0, 0, 0, 1, 1.f, 0, 1);
  concat_kernel<<<(int)(((long long)NROWS * (ENC_N + AA_N) + 255) / 256), 256, 0, stream>>>(
      u2f, aa, cat_h, (long long)NROWS * (ENC_N + AA_N));
  gemm(cat_h, out_wh, out_b, z0f, NROWS, OUT_N, ENC_N + AA_N, ENC_N + AA_N, OUT_N, OUT_N,
       0, 0, 0, 1, 1.f, 0, 0);
  tailact_kernel<<<(NROWS * OUT_N + 255) / 256, 256, 0, stream>>>(
      z0f, zf, out_dyt_a, out_dyt_w, out_dyt_b, (long long)NROWS * OUT_N);
  vq_kernel<<<(NROWS + 255) / 256, 256, 0, stream>>>(zf, codebook, (float*)d_out, vqacc);
  loss_kernel<<<1, 32, 0, stream>>>((float*)d_out + (long long)NROWS * OUT_N, vqacc);
}
